// RNNEventGenerator_22797686407499
// MI455X (gfx1250) — matrix operand order repaired and checked by reading, not compiled, not run on hardware
//
#include <hip/hip_runtime.h>
#include <hip/hip_bf16.h>
#include <math.h>

// ---- problem constants (match reference) ----
#define B_     32
#define CPD    64
#define T_     1024
#define IN_DIM 256
#define HID    512
#define WIN    1024

typedef __attribute__((ext_vector_type(16))) __bf16 v16bf;
typedef __attribute__((ext_vector_type(8)))  __bf16 v8bf;
typedef __attribute__((ext_vector_type(8)))  float  v8f;

// TDM descriptor vector types (per CDNA5_HIP.md probe)
typedef __attribute__((ext_vector_type(4))) unsigned int u32x4;
typedef __attribute__((ext_vector_type(4))) int          i32x4;
typedef __attribute__((ext_vector_type(8))) int          i32x8;

// ---- workspace layout (~98 MB total) ----
static constexpr size_t OFF_P    = 0;                                  // CPD*HID  bf16, K-major
static constexpr size_t OFF_WHH  = OFF_P    + (size_t)CPD*HID*2;       // HID*HID  bf16, K-major (W_hh^T)
static constexpr size_t OFF_WOUT = OFF_WHH  + (size_t)HID*HID*2;       // HID*WIN  bf16, K-major (W_out^T)
static constexpr size_t OFF_XW   = OFF_WOUT + (size_t)HID*WIN*2;       // B*T*HID  f32
static constexpr size_t OFF_HS   = OFF_XW   + (size_t)B_*T_*HID*4;     // B*T*HID  bf16

// ====================================================================
// Prep: P = proj @ W_ih^T  (64x512, K=256) stored K-major bf16
// ====================================================================
__global__ void k_prep_p(const float* __restrict__ proj,
                         const float* __restrict__ W_ih,
                         __bf16* __restrict__ P) {
  int idx = blockIdx.x * blockDim.x + threadIdx.x;
  if (idx >= CPD * HID) return;
  int c = idx / HID, h = idx % HID;
  const float* pr = proj + (size_t)c * IN_DIM;
  const float* wi = W_ih + (size_t)h * IN_DIM;
  float acc = 0.f;
  #pragma unroll 8
  for (int i = 0; i < IN_DIM; ++i) acc += pr[i] * wi[i];
  P[(size_t)c * HID + h] = (__bf16)acc;
}

__global__ void k_prep_whh(const float* __restrict__ W_hh, __bf16* __restrict__ Wk) {
  int idx = blockIdx.x * blockDim.x + threadIdx.x;
  if (idx >= HID * HID) return;
  int k = idx / HID, n = idx % HID;
  Wk[idx] = (__bf16)W_hh[(size_t)n * HID + k];
}

__global__ void k_prep_wout(const float* __restrict__ W_out, __bf16* __restrict__ Wk) {
  int idx = blockIdx.x * blockDim.x + threadIdx.x;
  if (idx >= HID * WIN) return;
  int h = idx / WIN, w = idx % WIN;
  Wk[idx] = (__bf16)W_out[(size_t)w * HID + h];
}

// ====================================================================
// Stage 1: xw(b,t,h) = sum_c control(b,c,t) * P(c,h)   K=64 (2 WMMA steps)
// Block = one (b, t-tile); 8 waves cover 8 adjacent h-tiles and share one
// LDS-staged A tile (64 c-rows x 16 t-cols, transposed to t-major bf16).
// ====================================================================
__global__ void __launch_bounds__(256) k_xw(const float* __restrict__ control,
                                            const __bf16* __restrict__ P,
                                            float* __restrict__ xw) {
  __shared__ __bf16 a_lds[16 * CPD];   // a_lds[m*64 + k] = control[b, k, t0+m]
  int tid  = threadIdx.x;
  int wave = tid >> 5;
  int lane = tid & 31;
  int g = lane >> 4, lr = lane & 15;

  const int TM = T_ / 16;              // 64
  int bid = blockIdx.x;
  int tn8 = bid & 3;  bid >>= 2;       // 4 groups of 8 h-tiles
  int tm  = bid % TM; int b = bid / TM;
  int t0 = tm * 16;
  int n0 = (tn8 * 8 + wave) * 16;

  // cooperative coalesced stage of the control tile (1024 f32)
  for (int i = tid; i < 16 * CPD; i += 256) {
    int col = i & 15;                  // t offset: adjacent threads -> adjacent t
    int c   = i >> 4;
    a_lds[col * CPD + c] = (__bf16)control[((size_t)b * CPD + c) * T_ + t0 + col];
  }
  __syncthreads();

  v8f acc = {};
  #pragma unroll
  for (int s = 0; s < 2; ++s) {
    // B fragment: lane l holds col n0 + l%16 of P, k rows in table order
    v16bf bfrag;
    #pragma unroll
    for (int i = 0; i < 8; ++i) { bfrag[i] = P[(size_t)(32 * s + 8 * g + i) * HID + n0 + lr]; bfrag[i + 8] = P[(size_t)(32 * s + 16 + 8 * g + i) * HID + n0 + lr]; }
    // A fragment from LDS: row m = lr, contiguous k -> two 16B ds loads
    const __bf16* ar = &a_lds[lr * CPD + 32 * s];
    v8bf lo = *(const v8bf*)(ar + 8 * g);
    v8bf hi = *(const v8bf*)(ar + 16 + 8 * g);
    v16bf afrag;
    #pragma unroll
    for (int i = 0; i < 8; ++i) { afrag[i] = lo[i]; afrag[i + 8] = hi[i]; }
    acc = __builtin_amdgcn_wmma_f32_16x16x32_bf16(false, afrag, false, bfrag,
                                                  (short)0, acc, false, false);
  }
  #pragma unroll
  for (int r = 0; r < 8; ++r) {
    int m = r + 8 * g;                 // t row within tile
    xw[((size_t)b * T_ + t0 + m) * HID + n0 + lr] = acc[r];
  }
}

// ====================================================================
// Stage 2: sequential recurrence  h = tanh(xw_t + h @ W_hh^T)
// One persistent workgroup (32 waves on one WGP).
// LDS (288 KB of the WGP's 320 KB):
//   [0, 256KB)   W_hh^T rows k=0..255, bf16 K-major  (loaded once via TDM)
//   [256, 288KB) h state, 32 x 512 bf16
// Per step: k-rows 0..255 of B come from LDS (ds), rows 256..511 stream
// from L2 (256 KB/step instead of 512 KB/step -> balanced with WMMA time).
// ====================================================================
__global__ void __launch_bounds__(1024) k_rnn(const float* __restrict__ xw,
                                              const __bf16* __restrict__ Whh,
                                              __bf16* __restrict__ hs) {
  extern __shared__ __align__(128) char smem_raw[];
  __bf16* whh_lds = (__bf16*)smem_raw;                      // 256 KB
  __bf16* h_lds   = (__bf16*)(smem_raw + 256 * 1024);       // 32 KB

  int tid = threadIdx.x;
  int wave = tid >> 5, lane = tid & 31, g = lane >> 4, lr = lane & 15;
  int n0 = wave * 16;

  for (int i = tid; i < B_ * HID; i += blockDim.x) h_lds[i] = (__bf16)0.f;

  // ---- TDM: bulk-copy W_hh^T rows 0..255 (256 KB) into LDS, wave 0 issues ----
  if (tid < 32) {
    unsigned long long ga = (unsigned long long)(uintptr_t)Whh;
    unsigned int       la = (unsigned int)(uintptr_t)whh_lds;
    u32x4 g0 = {};                       // D# group 0
    g0[0] = 1u;                          // count=1 (valid user descriptor)
    g0[1] = la;                          // lds_addr (bytes)
    g0[2] = (unsigned int)ga;            // global_addr[31:0]
    g0[3] = (unsigned int)((ga >> 32) & 0x01FFFFFFu) | (2u << 30); // [56:32] | type=2
    i32x8 g1 = {};                       // D# group 1
    g1[0] = (int)(2u << 16);             // data_size=2 -> 4-byte elements; wg_mask=0
    g1[1] = (int)(256u << 16);           // tensor_dim0 lo16 (256 dwords/row)
    g1[2] = (int)(256u << 16);           // tensor_dim1 lo16 (256 rows)
    g1[3] = (int)(256u << 16);           // tile_dim0 = 256 dwords
    g1[4] = (int)256;                    // tile_dim1 = 256 rows
    g1[5] = (int)256;                    // tensor_dim0_stride = 256 dwords
    i32x4 g2 = {}, g3 = {};              // 2D tensor: groups 2/3 zero
#if __has_include(<hip/amd_detail/amd_gfx1250_TDM.h>)
    i32x8 g4 = {};
    __builtin_amdgcn_tensor_load_to_lds(g0, g1, g2, g3, g4, 0);   // 6-arg toolchain
#else
    __builtin_amdgcn_tensor_load_to_lds(g0, g1, g2, g3, 0);       // 5-arg ROCm 7.2
#endif
    __builtin_amdgcn_s_wait_tensorcnt(0);
  }
  __syncthreads();

  #pragma unroll 1
  for (int t = 0; t < T_; ++t) {
    // C = xw(:, t, n0+lr) in f32 (full-precision bias in the accumulator)
    v8f acc0 = {}, acc1 = {};
    #pragma unroll
    for (int r = 0; r < 8; ++r) {
      int b0 = r + 8 * g;
      acc0[r] = xw[((size_t)b0        * T_ + t) * HID + n0 + lr];
      acc1[r] = xw[((size_t)(b0 + 16) * T_ + t) * HID + n0 + lr];
    }
    // k-rows 0..255: B from LDS
    #pragma unroll 4
    for (int s = 0; s < 8; ++s) {
      v16bf bfrag;
      #pragma unroll
      for (int i = 0; i < 8; ++i) { bfrag[i] = whh_lds[(size_t)(32 * s + 8 * g + i) * HID + n0 + lr]; bfrag[i + 8] = whh_lds[(size_t)(32 * s + 16 + 8 * g + i) * HID + n0 + lr]; }
      const __bf16* r0 = &h_lds[(size_t)lr        * HID + 32 * s];
      const __bf16* r1 = &h_lds[(size_t)(lr + 16) * HID + 32 * s];
      v16bf a0, a1;
      { v8bf lo = *(const v8bf*)(r0 + 8 * g); v8bf hi = *(const v8bf*)(r0 + 16 + 8 * g);
        #pragma unroll
        for (int i = 0; i < 8; ++i) { a0[i] = lo[i]; a0[i + 8] = hi[i]; } }
      { v8bf lo = *(const v8bf*)(r1 + 8 * g); v8bf hi = *(const v8bf*)(r1 + 16 + 8 * g);
        #pragma unroll
        for (int i = 0; i < 8; ++i) { a1[i] = lo[i]; a1[i + 8] = hi[i]; } }
      acc0 = __builtin_amdgcn_wmma_f32_16x16x32_bf16(false, a0, false, bfrag,
                                                     (short)0, acc0, false, false);
      acc1 = __builtin_amdgcn_wmma_f32_16x16x32_bf16(false, a1, false, bfrag,
                                                     (short)0, acc1, false, false);
    }
    // k-rows 256..511: B streamed from L2-resident second half
    #pragma unroll 4
    for (int s = 8; s < 16; ++s) {
      v16bf bfrag;
      #pragma unroll
      for (int i = 0; i < 8; ++i) { bfrag[i] = Whh[(size_t)(32 * s + 8 * g + i) * HID + n0 + lr]; bfrag[i + 8] = Whh[(size_t)(32 * s + 16 + 8 * g + i) * HID + n0 + lr]; }
      const __bf16* r0 = &h_lds[(size_t)lr        * HID + 32 * s];
      const __bf16* r1 = &h_lds[(size_t)(lr + 16) * HID + 32 * s];
      v16bf a0, a1;
      { v8bf lo = *(const v8bf*)(r0 + 8 * g); v8bf hi = *(const v8bf*)(r0 + 16 + 8 * g);
        #pragma unroll
        for (int i = 0; i < 8; ++i) { a0[i] = lo[i]; a0[i + 8] = hi[i]; } }
      { v8bf lo = *(const v8bf*)(r1 + 8 * g); v8bf hi = *(const v8bf*)(r1 + 16 + 8 * g);
        #pragma unroll
        for (int i = 0; i < 8; ++i) { a1[i] = lo[i]; a1[i + 8] = hi[i]; } }
      acc0 = __builtin_amdgcn_wmma_f32_16x16x32_bf16(false, a0, false, bfrag,
                                                     (short)0, acc0, false, false);
      acc1 = __builtin_amdgcn_wmma_f32_16x16x32_bf16(false, a1, false, bfrag,
                                                     (short)0, acc1, false, false);
    }
    __syncthreads();  // everyone done reading old h
    #pragma unroll
    for (int r = 0; r < 8; ++r) {
      int b0 = r + 8 * g;
      __bf16 h0 = (__bf16)tanhf(acc0[r]);
      __bf16 h1 = (__bf16)tanhf(acc1[r]);
      h_lds[(size_t)b0        * HID + n0 + lr] = h0;
      h_lds[(size_t)(b0 + 16) * HID + n0 + lr] = h1;
      hs[((size_t)b0        * T_ + t) * HID + n0 + lr] = h0;
      hs[((size_t)(b0 + 16) * T_ + t) * HID + n0 + lr] = h1;
    }
    __syncthreads();  // new h visible before next step
  }
}

// ====================================================================
// Stage 3: out(row, w) = sin( sum_h hs(row, h) * W_out(w, h) )
// rows = b*T + t (32768), K=512 -> 16 WMMA steps per 16x16 tile.
// ====================================================================
__global__ void __launch_bounds__(256) k_out(const __bf16* __restrict__ hs,
                                             const __bf16* __restrict__ Wout,
                                             float* __restrict__ out) {
  int wave = threadIdx.x >> 5;
  int lane = threadIdx.x & 31;
  int g = lane >> 4, lr = lane & 15;
  int id = blockIdx.x * (blockDim.x >> 5) + wave;
  const int TN = WIN / 16;                 // 64
  int tn = id % TN, tm = id / TN;          // tm < 2048
  if (tm >= (B_ * T_) / 16) return;
  int m0 = tm * 16, n0 = tn * 16;

  v8f acc = {};
  const __bf16* arow = hs + (size_t)(m0 + lr) * HID;  // row-major, contiguous k
  #pragma unroll 4
  for (int s = 0; s < 16; ++s) {
    v16bf bfrag;
    #pragma unroll
    for (int i = 0; i < 8; ++i) { bfrag[i] = Wout[(size_t)(32 * s + 8 * g + i) * WIN + n0 + lr]; bfrag[i + 8] = Wout[(size_t)(32 * s + 16 + 8 * g + i) * WIN + n0 + lr]; }
    v8bf lo = *(const v8bf*)(arow + 32 * s + 8 * g);
    v8bf hi = *(const v8bf*)(arow + 32 * s + 16 + 8 * g);
    v16bf a;
    #pragma unroll
    for (int i = 0; i < 8; ++i) { a[i] = lo[i]; a[i + 8] = hi[i]; }
    acc = __builtin_amdgcn_wmma_f32_16x16x32_bf16(false, a, false, bfrag,
                                                  (short)0, acc, false, false);
  }
  #pragma unroll
  for (int r = 0; r < 8; ++r) {
    int m = m0 + r + 8 * g;
    out[(size_t)m * WIN + n0 + lr] = __sinf(acc[r]);
  }
}

// ====================================================================
extern "C" void kernel_launch(void* const* d_in, const int* in_sizes, int n_in,
                              void* d_out, int out_size, void* d_ws, size_t ws_size,
                              hipStream_t stream) {
  const float* control = (const float*)d_in[0];  // (B, CPD, T)
  const float* proj    = (const float*)d_in[1];  // (CPD, IN_DIM)
  const float* W_ih    = (const float*)d_in[2];  // (HID, IN_DIM)
  const float* W_hh    = (const float*)d_in[3];  // (HID, HID)
  const float* W_out   = (const float*)d_in[4];  // (WIN, HID)

  char* ws = (char*)d_ws;   // requires ~98 MB workspace
  __bf16* P    = (__bf16*)(ws + OFF_P);
  __bf16* Whh  = (__bf16*)(ws + OFF_WHH);
  __bf16* Wout = (__bf16*)(ws + OFF_WOUT);
  float*  xw   = (float*) (ws + OFF_XW);
  __bf16* hs   = (__bf16*)(ws + OFF_HS);

  k_prep_p   <<<(CPD * HID + 255) / 256, 256, 0, stream>>>(proj, W_ih, P);
  k_prep_whh <<<(HID * HID + 255) / 256, 256, 0, stream>>>(W_hh, Whh);
  k_prep_wout<<<(HID * WIN + 255) / 256, 256, 0, stream>>>(W_out, Wout);

  // blocks = B * (T/16) * (HID/(16*8)) = 32*64*4 = 8192; 8 waves share an A tile
  k_xw<<<8192, 256, 0, stream>>>(control, P, xw);

  // single persistent workgroup, 288 KB dynamic LDS (W_hh^T half + h state)
  k_rnn<<<1, 1024, (256 + 32) * 1024, stream>>>(xw, Whh, hs);

  // (32768/16) * (1024/16) = 131072 wave-tiles, 8 waves per block
  k_out<<<131072 / 8, 256, 0, stream>>>(hs, Wout, (float*)d_out);
}
